// SimVQ_63127429316660
// MI455X (gfx1250) — matrix operand order repaired and checked by reading, not compiled, not run on hardware
//
#include <hip/hip_runtime.h>

// ---------------------------------------------------------------------------
// SimVQ for MI455X (gfx1250, wave32, WMMA).
// Dominant work: 16384x8192x512 distance GEMM + argmin, done as 3x bf16
// WMMA (hi*hi + hi*lo + lo*hi split) with f32 accumulation => ~exact argmin.
// This revision: double-buffered LDS staging (one barrier per K-step,
// global->LDS copies overlap the WMMA stream).
// ---------------------------------------------------------------------------

typedef __attribute__((ext_vector_type(16))) __bf16        v16bf;
typedef __attribute__((ext_vector_type(8)))  float         v8f;
typedef __attribute__((ext_vector_type(4)))  unsigned int  u32x4;

#define DIMD   512
#define KPAIR  (DIMD / 2)      // 256 packed bf16-pair dwords per row
#define CSIZE  8192
#define NROWS  16384           // B*N = 8*2048
#define NSPLIT 8               // candidate-dim grid split
#define QOFF   (NROWS * DIMD)  // d_out offset of indices region
#define PITCH  24              // LDS tile pitch (dwords): 32B-aligned frags
#define TBUF   (4 * 128 * PITCH)  // one tile buffer (A hi/lo + B hi/lo)

// ---- bf16 split helpers (round-to-nearest-even) ---------------------------
static __device__ __forceinline__ unsigned short f2bf(float f) {
  unsigned int u = __builtin_bit_cast(unsigned int, f);
  u += 0x7FFFu + ((u >> 16) & 1u);
  return (unsigned short)(u >> 16);
}
static __device__ __forceinline__ float bf2f(unsigned short h) {
  unsigned int u = ((unsigned int)h) << 16;
  return __builtin_bit_cast(float, u);
}
static __device__ __forceinline__ unsigned int pack_hi(float a, float b) {
  return (unsigned int)f2bf(a) | ((unsigned int)f2bf(b) << 16);
}
static __device__ __forceinline__ unsigned int pack_lo(float a, float b) {
  const unsigned short ha = f2bf(a), hb = f2bf(b);
  return (unsigned int)f2bf(a - bf2f(ha)) | ((unsigned int)f2bf(b - bf2f(hb)) << 16);
}

// ---- WMMA wrapper (codegen-confirmed builtin) ------------------------------
static __device__ __forceinline__ v8f wmma_bf16(v16bf a, v16bf b, v8f c) {
  return __builtin_amdgcn_wmma_f32_16x16x32_bf16(
      false, a, false, b, (short)0, c, false, false);
}
// Fragment = 8 contiguous dwords (32B aligned) in LDS -> 2x ds_load_b128.
static __device__ __forceinline__ v16bf load_frag(const unsigned int* p) {
  return *(const v16bf*)p;
}

// ===========================================================================
// Kernel 1: codes = codebook @ W^T (f32, tiled). Outputs:
//   codes  [c][e]        f32 (exact, for gather/rotation)
//   cpk_hi [e/2][c] u32  packed bf16 pair (e even | e odd << 16)
//   cpk_lo [e/2][c] u32  packed bf16 residual pair
// ===========================================================================
__global__ __launch_bounds__(256) void k_codes(
    const float* __restrict__ cb, const float* __restrict__ W,
    float* __restrict__ codes, unsigned int* __restrict__ cpk_hi,
    unsigned int* __restrict__ cpk_lo) {
  __shared__ float As[64][17];
  __shared__ float Bs[16][65];
  const int tid = threadIdx.x;
  const int tx = tid & 15, ty = tid >> 4;
  const int c0 = blockIdx.x * 64;  // candidate tile
  const int e0 = blockIdx.y * 64;  // feature tile
  float acc[4][4] = {};

  for (int kt = 0; kt < DIMD / 16; ++kt) {
    const int k0 = kt * 16;
#pragma unroll
    for (int i = 0; i < 4; ++i) {
      const int e = tid + i * 256;
      As[e >> 4][e & 15] = cb[(size_t)(c0 + (e >> 4)) * DIMD + k0 + (e & 15)];
    }
#pragma unroll
    for (int i = 0; i < 4; ++i) {
      const int e = tid + i * 256;
      Bs[e & 15][e >> 4] = W[(size_t)(e0 + (e >> 4)) * DIMD + k0 + (e & 15)];
    }
    __syncthreads();
#pragma unroll
    for (int k = 0; k < 16; ++k)
#pragma unroll
      for (int i = 0; i < 4; ++i) {
        const float a = As[ty * 4 + i][k];
#pragma unroll
        for (int j = 0; j < 4; ++j) acc[i][j] += a * Bs[k][tx * 4 + j];
      }
    __syncthreads();
  }
  const int e_base = e0 + tx * 4;
  const int kk0 = e_base >> 1;  // pair index (even)
#pragma unroll
  for (int i = 0; i < 4; ++i) {
    const int c = c0 + ty * 4 + i;
#pragma unroll
    for (int j = 0; j < 4; ++j) codes[(size_t)c * DIMD + e_base + j] = acc[i][j];
    cpk_hi[(size_t)kk0 * CSIZE + c]       = pack_hi(acc[i][0], acc[i][1]);
    cpk_hi[(size_t)(kk0 + 1) * CSIZE + c] = pack_hi(acc[i][2], acc[i][3]);
    cpk_lo[(size_t)kk0 * CSIZE + c]       = pack_lo(acc[i][0], acc[i][1]);
    cpk_lo[(size_t)(kk0 + 1) * CSIZE + c] = pack_lo(acc[i][2], acc[i][3]);
  }
}

// ===========================================================================
// Kernel 2: pack x into bf16 hi/lo pair dwords: xpk[row][kk], kk = e/2.
// (Stored temporarily in the d_out quantized region; overwritten by k_rotate.)
// ===========================================================================
__global__ __launch_bounds__(256) void k_xsplit(
    const float* __restrict__ x, unsigned int* __restrict__ xpk_hi,
    unsigned int* __restrict__ xpk_lo) {
  const size_t idx = (size_t)blockIdx.x * 256 + threadIdx.x;  // pair index
  const float2 v = *(const float2*)(x + idx * 2);
  xpk_hi[idx] = pack_hi(v.x, v.y);
  xpk_lo[idx] = pack_lo(v.x, v.y);
}

// ===========================================================================
// Kernel 3: codesq[c] = sum_e codes[c][e]^2  (one wave per row)
// ===========================================================================
__global__ __launch_bounds__(256) void k_codesq(const float* __restrict__ codes,
                                                float* __restrict__ codesq) {
  const int wave = threadIdx.x >> 5, lane = threadIdx.x & 31;
  const int row = blockIdx.x * 8 + wave;
  const float* cr = codes + (size_t)row * DIMD;
  float s = 0.f;
  for (int i = lane; i < DIMD; i += 32) s += cr[i] * cr[i];
#pragma unroll
  for (int off = 16; off >= 1; off >>= 1) s += __shfl_xor(s, off);
  if (lane == 0) codesq[row] = s;
}

// ===========================================================================
// Kernel 4: WMMA distance + running argmin, double-buffered LDS staging.
//   Grid: 128 row-blocks x NSPLIT cand-splits. WG=256 (8 waves).
//   WG tile: 128 rows x 128 cand per chunk; wave = 32 rows x 64 cand.
//   LDS tiles in *fragment order*:
//     A: [row][kpair]; lane frag = 8 dwords at
//        row*PITCH + h*8  (K0 = 2j+16h, same order as B)
//     B: [n][kpair]; lane frag = 8 dwords at n*PITCH + h*8 (K0 = 2j+16h)
// ===========================================================================
static __device__ __forceinline__ void stage_tiles(
    unsigned int* __restrict__ sm, const unsigned int* __restrict__ xpk_hi,
    const unsigned int* __restrict__ xpk_lo,
    const unsigned int* __restrict__ cpk_hi,
    const unsigned int* __restrict__ cpk_lo, int tid, int R0, int Cc, int kk0) {
  unsigned int* Al_hi = sm;
  unsigned int* Al_lo = sm + 128 * PITCH;
  unsigned int* Bl_hi = sm + 2 * 128 * PITCH;
  unsigned int* Bl_lo = sm + 3 * 128 * PITCH;
  // A (x) tile: 128 rows x 16 pairs, u32x4 copies (contiguous pair order)
#pragma unroll
  for (int i = 0; i < 2; ++i) {
    const int g = tid + i * 256;  // 512 quad-groups
    const int row = g >> 2, q = g & 3;
    const size_t ga = (size_t)(R0 + row) * KPAIR + kk0 + q * 4;
    const u32x4 vh = *(const u32x4*)(xpk_hi + ga);
    const u32x4 vl = *(const u32x4*)(xpk_lo + ga);
    const int dq = q;
    *(u32x4*)(Al_hi + row * PITCH + dq * 4) = vh;
    *(u32x4*)(Al_lo + row * PITCH + dq * 4) = vl;
  }
  // B (codes) tile: 16 pairs x 128 cand
#pragma unroll
  for (int i = 0; i < 2; ++i) {
    const int g = tid + i * 256;
    const int jj = g >> 5, n4 = (g & 31) * 4;
    const size_t gb = (size_t)(kk0 + jj) * CSIZE + Cc + n4;
    const u32x4 vh = *(const u32x4*)(cpk_hi + gb);
    const u32x4 vl = *(const u32x4*)(cpk_lo + gb);
#pragma unroll
    for (int t = 0; t < 4; ++t) {
      Bl_hi[(n4 + t) * PITCH + jj] = vh[t];
      Bl_lo[(n4 + t) * PITCH + jj] = vl[t];
    }
  }
}

__global__ __launch_bounds__(256) void k_dist(
    const unsigned int* __restrict__ xpk_hi, const unsigned int* __restrict__ xpk_lo,
    const unsigned int* __restrict__ cpk_hi, const unsigned int* __restrict__ cpk_lo,
    const float* __restrict__ codesq, float* __restrict__ part_val,
    int* __restrict__ part_idx) {
  __shared__ __align__(64) unsigned int smem[2 * TBUF];  // 96 KB, ping-pong

  const int tid = threadIdx.x;
  const int wave = tid >> 5, lane = tid & 31;
  const int l = lane & 15, h = lane >> 4;
  const int wm = wave >> 1;  // 0..3 row group (32 rows)
  const int wn = wave & 1;   // 0..1 cand group (64 cands)

  const int wg_m = blockIdx.x & 127;
  const int wg_n = blockIdx.x >> 7;        // 0..NSPLIT-1
  const int R0 = wg_m * 128;
  const int C0 = wg_n * (CSIZE / NSPLIT);  // 1024 candidates per split

  float minv[2][8];
  int mini[2][8];
#pragma unroll
  for (int mi = 0; mi < 2; ++mi)
#pragma unroll
    for (int r = 0; r < 8; ++r) { minv[mi][r] = 3.402823466e38f; mini[mi][r] = 0; }

  for (int chunk = 0; chunk < (CSIZE / NSPLIT) / 128; ++chunk) {
    const int Cc = C0 + chunk * 128;
    v8f acc[2][4];
    const v8f vz = {0.f, 0.f, 0.f, 0.f, 0.f, 0.f, 0.f, 0.f};
#pragma unroll
    for (int mi = 0; mi < 2; ++mi)
#pragma unroll
      for (int ni = 0; ni < 4; ++ni) acc[mi][ni] = vz;

    // prologue: stage K-step 0 into buffer 0
    stage_tiles(smem, xpk_hi, xpk_lo, cpk_hi, cpk_lo, tid, R0, Cc, 0);
    __syncthreads();

    for (int ks = 0; ks < DIMD / 32; ++ks) {
      // stage next K-step into the idle buffer (overlaps with WMMA below)
      if (ks + 1 < DIMD / 32)
        stage_tiles(smem + ((ks + 1) & 1) * TBUF, xpk_hi, xpk_lo, cpk_hi,
                    cpk_lo, tid, R0, Cc, (ks + 1) * 16);

      const unsigned int* sm = smem + (ks & 1) * TBUF;
      const unsigned int* Al_hi = sm;
      const unsigned int* Al_lo = sm + 128 * PITCH;
      const unsigned int* Bl_hi = sm + 2 * 128 * PITCH;
      const unsigned int* Bl_lo = sm + 3 * 128 * PITCH;

      const v16bf ah0 = load_frag(Al_hi + (wm * 32 + l) * PITCH + h * 8);
      const v16bf ah1 = load_frag(Al_hi + (wm * 32 + 16 + l) * PITCH + h * 8);
      const v16bf al0 = load_frag(Al_lo + (wm * 32 + l) * PITCH + h * 8);
      const v16bf al1 = load_frag(Al_lo + (wm * 32 + 16 + l) * PITCH + h * 8);
#pragma unroll
      for (int ni = 0; ni < 4; ++ni) {
        const int n0 = wn * 64 + ni * 16;
        const v16bf bh = load_frag(Bl_hi + (n0 + l) * PITCH + h * 8);
        const v16bf bl = load_frag(Bl_lo + (n0 + l) * PITCH + h * 8);
        acc[0][ni] = wmma_bf16(ah0, bh, acc[0][ni]);
        acc[0][ni] = wmma_bf16(ah0, bl, acc[0][ni]);
        acc[0][ni] = wmma_bf16(al0, bh, acc[0][ni]);
        acc[1][ni] = wmma_bf16(ah1, bh, acc[1][ni]);
        acc[1][ni] = wmma_bf16(ah1, bl, acc[1][ni]);
        acc[1][ni] = wmma_bf16(al1, bh, acc[1][ni]);
      }
      // one barrier per step: staging of ks+1 visible, buffer ks reusable
      __syncthreads();
    }
    // fold chunk into running per-lane argmin.  D layout: M = r+8h, N = l.
#pragma unroll
    for (int ni = 0; ni < 4; ++ni) {
      const int cg = Cc + wn * 64 + ni * 16 + l;
      const float cq = codesq[cg];
#pragma unroll
      for (int mi = 0; mi < 2; ++mi)
#pragma unroll
        for (int r = 0; r < 8; ++r) {
          const float d = cq - 2.0f * acc[mi][ni][r];
          if (d < minv[mi][r]) { minv[mi][r] = d; mini[mi][r] = cg; }
        }
    }
  }
  // reduce across the 16 lanes of each half (tie -> lower index)
#pragma unroll
  for (int mi = 0; mi < 2; ++mi)
#pragma unroll
    for (int r = 0; r < 8; ++r) {
      float v = minv[mi][r];
      int id = mini[mi][r];
      for (int off = 8; off >= 1; off >>= 1) {
        const float ov = __shfl_xor(v, off);
        const int oi = __shfl_xor(id, off);
        if (ov < v || (ov == v && oi < id)) { v = ov; id = oi; }
      }
      minv[mi][r] = v; mini[mi][r] = id;
    }
  __syncthreads();
  // merge the two wn waves per row via LDS, write partials
  float* mv = (float*)smem;       // [128][2]
  int* mix = (int*)(smem + 256);  // [128][2]
  if (l == 0) {
#pragma unroll
    for (int mi = 0; mi < 2; ++mi)
#pragma unroll
      for (int r = 0; r < 8; ++r) {
        const int lr = wm * 32 + mi * 16 + 8 * h + r;
        mv[lr * 2 + wn] = minv[mi][r];
        mix[lr * 2 + wn] = mini[mi][r];
      }
  }
  __syncthreads();
  if (tid < 128) {
    const float v0 = mv[tid * 2], v1 = mv[tid * 2 + 1];
    const int i0 = mix[tid * 2], i1 = mix[tid * 2 + 1];
    const bool t1 = (v1 < v0) || (v1 == v0 && i1 < i0);
    const int row = R0 + tid;
    part_val[row * NSPLIT + wg_n] = t1 ? v1 : v0;
    part_idx[row * NSPLIT + wg_n] = t1 ? i1 : i0;
  }
}

// ===========================================================================
// Kernel 5: merge candidate splits -> final index; emit indices as f32
// ===========================================================================
__global__ __launch_bounds__(256) void k_merge(
    const float* __restrict__ part_val, const int* __restrict__ part_idx,
    int* __restrict__ final_idx, float* __restrict__ out_idx_f) {
  const int row = blockIdx.x * 256 + threadIdx.x;
  float bv = part_val[row * NSPLIT];
  int bi = part_idx[row * NSPLIT];
#pragma unroll
  for (int s = 1; s < NSPLIT; ++s) {
    const float v = part_val[row * NSPLIT + s];
    const int i = part_idx[row * NSPLIT + s];
    if (v < bv || (v == bv && i < bi)) { bv = v; bi = i; }
  }
  final_idx[row] = bi;
  out_idx_f[row] = (float)bi;
}

// ===========================================================================
// Kernel 6: gather + rotation trick (forward == q, computed exactly like the
// reference formula) + per-row commit partial.  One wave per row.
// ===========================================================================
__global__ __launch_bounds__(256) void k_rotate(
    const float* __restrict__ x, const float* __restrict__ codes,
    const int* __restrict__ final_idx, float* __restrict__ out_q,
    float* __restrict__ row_commit) {
  const int wave = threadIdx.x >> 5, lane = threadIdx.x & 31;
  const int row = blockIdx.x * 8 + wave;
  const float* xr = x + (size_t)row * DIMD;
  const float* qr = codes + (size_t)final_idx[row] * DIMD;

  float xv[16], qv[16];
  float sx = 0.f, sq = 0.f, sxq = 0.f, sd = 0.f;
#pragma unroll
  for (int i = 0; i < 16; ++i) {
    const float a = xr[lane + i * 32], b = qr[lane + i * 32];
    xv[i] = a; qv[i] = b;
    sx += a * a; sq += b * b; sxq += a * b;
    const float d = a - b; sd += d * d;
  }
#pragma unroll
  for (int off = 16; off >= 1; off >>= 1) {
    sx += __shfl_xor(sx, off);
    sq += __shfl_xor(sq, off);
    sxq += __shfl_xor(sxq, off);
    sd += __shfl_xor(sd, off);
  }
  const float nx = sqrtf(sx), nq = sqrtf(sq);
  const float nxm = fmaxf(nx, 1e-12f), nqm = fmaxf(nq, 1e-12f);
  float su = 0.f;
#pragma unroll
  for (int i = 0; i < 16; ++i) {
    const float u = xv[i] / nxm + qv[i] / nqm;
    su += u * u;
  }
#pragma unroll
  for (int off = 16; off >= 1; off >>= 1) su += __shfl_xor(su, off);
  const float nu = fmaxf(sqrtf(su), 1e-12f);
  const float ew = (sx / nxm + sxq / nqm) / nu;  // e.w
  const float ex = sx / nxm;                     // e.x_hat
  const float scale = nq / nxm;
#pragma unroll
  for (int i = 0; i < 16; ++i) {
    const float w = (xv[i] / nxm + qv[i] / nqm) / nu;
    const float qh = qv[i] / nqm;
    out_q[(size_t)row * DIMD + lane + i * 32] =
        (xv[i] - 2.f * ew * w + 2.f * ex * qh) * scale;
  }
  if (lane == 0) row_commit[row] = sd;
}

// ===========================================================================
// Kernel 7: deterministic reduction of per-row commit sums -> mean
// ===========================================================================
__global__ __launch_bounds__(256) void k_loss(const float* __restrict__ row_commit,
                                              float* __restrict__ out) {
  __shared__ float s[256];
  float p = 0.f;
  for (int i = threadIdx.x; i < NROWS; i += 256) p += row_commit[i];
  s[threadIdx.x] = p;
  __syncthreads();
  for (int st = 128; st > 0; st >>= 1) {
    if (threadIdx.x < st) s[threadIdx.x] += s[threadIdx.x + st];
    __syncthreads();
  }
  if (threadIdx.x == 0) out[0] = s[0] / (float)(NROWS * DIMD);
}

// ===========================================================================
extern "C" void kernel_launch(void* const* d_in, const int* in_sizes, int n_in,
                              void* d_out, int out_size, void* d_ws, size_t ws_size,
                              hipStream_t stream) {
  const float* x = (const float*)d_in[0];   // [8,2048,512]
  const float* cb = (const float*)d_in[1];  // [8192,512]
  const float* W = (const float*)d_in[2];   // [512,512]
  float* out = (float*)d_out;               // quantized | indices | loss

  // workspace carve (~34 MB)
  float* codes = (float*)d_ws;                                  // 8192*512 f32
  unsigned int* cpk_hi = (unsigned int*)(codes + CSIZE * DIMD); // 256*8192 u32
  unsigned int* cpk_lo = cpk_hi + (size_t)KPAIR * CSIZE;        // 256*8192 u32
  float* codesq = (float*)(cpk_lo + (size_t)KPAIR * CSIZE);     // 8192 f32
  float* part_val = codesq + CSIZE;                             // 16384*8 f32
  int* part_idx = (int*)(part_val + NROWS * NSPLIT);            // 16384*8 i32
  int* final_idx = part_idx + NROWS * NSPLIT;                   // 16384 i32
  float* row_commit = (float*)(final_idx + NROWS);              // 16384 f32

  // x hi/lo packed pairs live temporarily in the d_out quantized region
  // (exactly QOFF dwords); k_rotate overwrites it afterwards.
  unsigned int* xpk_hi = (unsigned int*)d_out;                  // 16384*256 u32
  unsigned int* xpk_lo = xpk_hi + (size_t)NROWS * KPAIR;        // 16384*256 u32

  k_codes<<<dim3(CSIZE / 64, DIMD / 64), 256, 0, stream>>>(cb, W, codes, cpk_hi, cpk_lo);
  k_xsplit<<<NROWS, 256, 0, stream>>>(x, xpk_hi, xpk_lo);
  k_codesq<<<CSIZE / 8, 256, 0, stream>>>(codes, codesq);
  k_dist<<<(NROWS / 128) * NSPLIT, 256, 0, stream>>>(xpk_hi, xpk_lo, cpk_hi, cpk_lo,
                                                     codesq, part_val, part_idx);
  k_merge<<<NROWS / 256, 256, 0, stream>>>(part_val, part_idx, final_idx, out + QOFF);
  k_rotate<<<NROWS / 8, 256, 0, stream>>>(x, codes, final_idx, out, row_commit);
  k_loss<<<1, 256, 0, stream>>>(row_commit, out + QOFF + NROWS);
}
